// LocalCurvatureGNN_58428735095026
// MI455X (gfx1250) — compile-verified, matrix operand order checked by reading, not run on hardware
//
#include <hip/hip_runtime.h>
#include <hip/hip_bf16.h>
#include <stdint.h>

typedef __attribute__((ext_vector_type(16))) _Float16 v16h;
typedef __attribute__((ext_vector_type(8)))  float    v8f;

#define M_IN  24
#define HID   32
#define OUTF  16
#define ZW    88   // 3*M + OUT
#define NW    10   // waves (16-node tiles) per node-kernel block

// ---------------- zero workspace accumulators ----------------
__global__ void gnn_zero(float* p, long long n) {
    long long i = (long long)blockIdx.x * blockDim.x + threadIdx.x;
    long long stride = (long long)gridDim.x * blockDim.x;
    for (; i < n; i += stride) p[i] = 0.0f;
}

// ---------------- edge scatter #1: feature-parallel ----------------
// one 32-lane group per edge; lane = feature -> contiguous coalesced
// gathers from x[row] and contiguous atomic adds into sumx[col].
__global__ void gnn_scatter1(const long long* __restrict__ ei,
                             const float* __restrict__ x,
                             float* __restrict__ deg,
                             float* __restrict__ sumx, int E) {
    long long tid = (long long)blockIdx.x * blockDim.x + threadIdx.x;
    int e = (int)(tid >> 5);
    int f = (int)(tid & 31);
    if (e >= E) return;
    int r = (int)ei[e];
    int c = (int)ei[(long long)E + e];
    if (f < M_IN) {
        atomicAdd(&sumx[(long long)c * M_IN + f], x[(long long)r * M_IN + f]);
    } else if (f == M_IN) {
        atomicAdd(&deg[c], 1.0f);
    }
}

// ---------------- edge scatter #2: feature-parallel (32 feats) ------------
__global__ void gnn_scatter2(const long long* __restrict__ ei,
                             const float* __restrict__ h,
                             float* __restrict__ sumh, int E) {
    long long tid = (long long)blockIdx.x * blockDim.x + threadIdx.x;
    int e = (int)(tid >> 5);
    int f = (int)(tid & 31);
    if (e >= E) return;
    int r = (int)ei[e];
    int c = (int)ei[(long long)E + e];
    atomicAdd(&sumh[(long long)c * HID + f], h[(long long)r * HID + f]);
}

// ---------------- layer 1: feats + SAGE1 (WMMA), NW waves/block -----------
// A = [frac1(24) 0pad | x(24) 0pad]  (16 x 64)
// W = [w1l | 0 | w1r | 0] as B tiles (32 x 64) ; h = relu(A @ W^T + b1)
__global__ __launch_bounds__(32 * NW) void gnn_node1(
    const float* __restrict__ x, const float* __restrict__ sumx,
    const float* __restrict__ deg,
    const float* __restrict__ w1l, const float* __restrict__ b1,
    const float* __restrict__ w1r, const float* __restrict__ headw,
    float* __restrict__ h, float* __restrict__ out, int N) {

    __shared__ _Float16 Ah[NW * 16 * 64];
    __shared__ _Float16 Wc[HID * 64];
    __shared__ float    b1s[HID];

    const int lane = threadIdx.x & 31;
    const int w    = threadIdx.x >> 5;
    const int base = (blockIdx.x * NW + w) * 16;
    const bool active = base < N;

    // stage combined weights into LDS as f16 (whole block cooperates)
    for (int idx = threadIdx.x; idx < HID * 64; idx += 32 * NW) {
        int o = idx >> 6, k = idx & 63;
        float v = 0.0f;
        if (k < M_IN)                      v = w1l[o * M_IN + k];
        else if (k >= 32 && k < 32 + M_IN) v = w1r[o * M_IN + (k - 32)];
        Wc[idx] = (_Float16)v;
    }
    if (threadIdx.x < HID) b1s[threadIdx.x] = b1[threadIdx.x];

    _Float16* A = Ah + w * (16 * 64);

    // per-node: frac1, hand features, head partial, A-tile
    if (active && lane < 16) {
        int node = base + lane;
        float d  = deg[node];
        d = d < 1.0f ? 1.0f : d;
        float inv = 1.0f / d;
        float yp  = 0.0f;
        float* z  = out + N + (long long)node * ZW;
#pragma unroll
        for (int f = 0; f < M_IN; ++f) {
            float xv = x[(long long)node * M_IN + f];
            float fr = sumx[(long long)node * M_IN + f] * inv;
            float pr = xv * fr;
            z[f]          = xv;
            z[M_IN + f]   = fr;
            z[2*M_IN + f] = pr;
            yp += xv * headw[f] + fr * headw[M_IN + f] + pr * headw[2*M_IN + f];
            A[lane * 64 + f]      = (_Float16)fr;
            A[lane * 64 + 32 + f] = (_Float16)xv;
        }
#pragma unroll
        for (int f = M_IN; f < 32; ++f) {
            A[lane * 64 + f]      = (_Float16)0.0f;
            A[lane * 64 + 32 + f] = (_Float16)0.0f;
        }
        out[node] = yp;  // partial yhat (feats part)
    }
    __syncthreads();

    if (active) {
        // A fragments (ISA 16-bit 16x32 layout)
        const int m  = lane & 15;
        const int hi = (lane >> 4) & 1;
        v16h a0, a1;
#pragma unroll
        for (int i = 0; i < 16; ++i) {
            int kk = ((i < 8) ? i : (i + 8)) + hi * 8;
            a0[i] = A[m * 64 + kk];
            a1[i] = A[m * 64 + 32 + kk];
        }
#pragma unroll
        for (int ct = 0; ct < 2; ++ct) {
            v16h bb0, bb1;
#pragma unroll
            for (int i = 0; i < 16; ++i) {
                int kk = ((i < 8) ? i : (i + 8)) + hi * 8;
                int o  = ct * 16 + m;
                bb0[i] = Wc[o * 64 + kk];
                bb1[i] = Wc[o * 64 + 32 + kk];
            }
            v8f c = {};
            c = __builtin_amdgcn_wmma_f32_16x16x32_f16(false, a0, false, bb0,
                                                       (short)0, c, false, false);
            c = __builtin_amdgcn_wmma_f32_16x16x32_f16(false, a1, false, bb1,
                                                       (short)0, c, false, false);
            int col  = ct * 16 + m;
            float bv = b1s[col];
#pragma unroll
            for (int r = 0; r < 8; ++r) {
                int node = base + r + hi * 8;
                float v  = c[r] + bv;
                v = v > 0.0f ? v : 0.0f;
                h[(long long)node * HID + col] = v;
            }
        }
    }
}

// ---------------- layer 2: SAGE2 (WMMA) + head, NW waves/block ------------
__global__ __launch_bounds__(32 * NW) void gnn_node2(
    const float* __restrict__ h, const float* __restrict__ sumh,
    const float* __restrict__ deg,
    const float* __restrict__ w2l, const float* __restrict__ b2,
    const float* __restrict__ w2r, const float* __restrict__ headw,
    const float* __restrict__ headb,
    float* __restrict__ out, int N) {

    __shared__ _Float16 Ah[NW * 16 * 64];
    __shared__ _Float16 Wc[OUTF * 64];
    __shared__ float    b2s[OUTF];
    __shared__ float    es[NW * 16 * 16];

    const int lane = threadIdx.x & 31;
    const int w    = threadIdx.x >> 5;
    const int base = (blockIdx.x * NW + w) * 16;
    const bool active = base < N;

    for (int idx = threadIdx.x; idx < OUTF * 64; idx += 32 * NW) {
        int o = idx >> 6, k = idx & 63;
        float v = (k < HID) ? w2l[o * HID + k] : w2r[o * HID + (k - HID)];
        Wc[idx] = (_Float16)v;
    }
    if (threadIdx.x < OUTF) b2s[threadIdx.x] = b2[threadIdx.x];

    _Float16* A  = Ah + w * (16 * 64);
    float*    ew = es + w * (16 * 16);

    if (active && lane < 16) {
        int node = base + lane;
        float d  = deg[node];
        d = d < 1.0f ? 1.0f : d;
        float inv = 1.0f / d;
#pragma unroll
        for (int f = 0; f < HID; ++f) {
            A[lane * 64 + f]      = (_Float16)(sumh[(long long)node * HID + f] * inv);
            A[lane * 64 + 32 + f] = (_Float16)h[(long long)node * HID + f];
        }
    }
    __syncthreads();

    const int m  = lane & 15;
    const int hi = (lane >> 4) & 1;
    if (active) {
        v16h a0, a1, bb0, bb1;
#pragma unroll
        for (int i = 0; i < 16; ++i) {
            int kk = ((i < 8) ? i : (i + 8)) + hi * 8;
            a0[i]  = A[m * 64 + kk];
            a1[i]  = A[m * 64 + 32 + kk];
            bb0[i] = Wc[m * 64 + kk];
            bb1[i] = Wc[m * 64 + 32 + kk];
        }
        v8f c = {};
        c = __builtin_amdgcn_wmma_f32_16x16x32_f16(false, a0, false, bb0,
                                                   (short)0, c, false, false);
        c = __builtin_amdgcn_wmma_f32_16x16x32_f16(false, a1, false, bb1,
                                                   (short)0, c, false, false);
#pragma unroll
        for (int r = 0; r < 8; ++r) {
            int node = base + r + hi * 8;
            float v  = c[r] + b2s[m];
            v = v > 0.0f ? v : 0.0f;
            out[(long long)N + (long long)node * ZW + 3*M_IN + m] = v;
            ew[(r + hi * 8) * 16 + m] = v;
        }
    }
    __syncthreads();

    if (active && lane < 16) {
        int node  = base + lane;
        float acc = out[node] + headb[0];
#pragma unroll
        for (int j = 0; j < OUTF; ++j) acc += ew[lane * 16 + j] * headw[3*M_IN + j];
        out[node] = acc;
    }
}

extern "C" void kernel_launch(void* const* d_in, const int* in_sizes, int n_in,
                              void* d_out, int out_size, void* d_ws, size_t ws_size,
                              hipStream_t stream) {
    const float*     x     = (const float*)d_in[0];
    const long long* ei    = (const long long*)d_in[1];
    const float*     w1l   = (const float*)d_in[2];
    const float*     b1    = (const float*)d_in[3];
    const float*     w1r   = (const float*)d_in[4];
    const float*     w2l   = (const float*)d_in[5];
    const float*     b2    = (const float*)d_in[6];
    const float*     w2r   = (const float*)d_in[7];
    const float*     headw = (const float*)d_in[8];
    const float*     headb = (const float*)d_in[9];
    float*           out   = (float*)d_out;

    const int N = in_sizes[0] / M_IN;       // 100000
    const int E = in_sizes[1] / 2;          // 3200000

    // workspace layout: deg[N] | sumx[N*24] | sumh[N*32] | h[N*32]
    float* deg  = (float*)d_ws;
    float* sumx = deg  + N;
    float* sumh = sumx + (long long)N * M_IN;
    float* h    = sumh + (long long)N * HID;

    const long long nzero = (long long)N * (1 + M_IN + HID);  // deg+sumx+sumh
    int zblocks = (int)((nzero + 255) / 256);
    if (zblocks > 65535) zblocks = 65535;
    gnn_zero<<<zblocks, 256, 0, stream>>>((float*)d_ws, nzero);

    // feature-parallel scatters: 32 lanes per edge
    long long sthreads = (long long)E * 32;
    int sblocks = (int)((sthreads + 255) / 256);
    gnn_scatter1<<<sblocks, 256, 0, stream>>>(ei, x, deg, sumx, E);

    int tiles   = (N + 15) / 16;
    int nblocks = (tiles + NW - 1) / NW;
    gnn_node1<<<nblocks, 32 * NW, 0, stream>>>(x, sumx, deg, w1l, b1, w1r,
                                               headw, h, out, N);

    gnn_scatter2<<<sblocks, 256, 0, stream>>>(ei, h, sumh, E);

    gnn_node2<<<nblocks, 32 * NW, 0, stream>>>(h, sumh, deg, w2l, b2, w2r,
                                               headw, headb, out, N);
}
